// Net_87952340287575
// MI455X (gfx1250) — matrix operand order repaired and checked by reading, not compiled, not run on hardware
//
#include <hip/hip_runtime.h>

typedef __attribute__((ext_vector_type(16))) _Float16 v16h;
typedef __attribute__((ext_vector_type(8)))  _Float16 v8h;
typedef __attribute__((ext_vector_type(8)))  float    v8f;
typedef __attribute__((ext_vector_type(4)))  float    v4f;

#define WPB 8   // waves per block (256 threads, wave32)

// ---------------------------------------------------------------- utilities
__global__ void zero_f32_kernel(float* __restrict__ p, long n) {
  long i = (long)blockIdx.x * blockDim.x + threadIdx.x;
  if (i < n) p[i] = 0.f;
}

// Convert W [NCo, Kin] f32 -> Wp [NCo, Kpad] f16 (zero-padded in K)
__global__ void wprep_kernel(const float* __restrict__ W, _Float16* __restrict__ Wp,
                             int Kin, int Kpad, int NCo) {
  int i = blockIdx.x * blockDim.x + threadIdx.x;
  if (i < Kpad * NCo) {
    int n = i / Kpad, k = i % Kpad;
    Wp[i] = (k < Kin) ? (_Float16)W[n * Kin + k] : (_Float16)0.f;
  }
}

// sum/sumsq -> scale/shift for BN:  y' = y*scale + shift
__global__ void finalize_kernel(const float* __restrict__ sum, const float* __restrict__ sumsq,
                                const float* __restrict__ g, const float* __restrict__ be,
                                float* __restrict__ scale, float* __restrict__ shift,
                                float invcnt, int NC) {
  int c = blockIdx.x * blockDim.x + threadIdx.x;
  if (c < NC) {
    float m   = sum[c] * invcnt;
    float var = sumsq[c] * invcnt - m * m;
    float rs  = rsqrtf(var + 1e-5f);
    float sc  = g[c] * rs;
    scale[c] = sc;
    shift[c] = be[c] - m * sc;
  }
}

// relu(bn(y)) scattered with max into xout[seg[r]] ; xout pre-zeroed, values>=0
// so uint-ordered atomicMax == float max, and empty segments stay 0.
__global__ void maxagg_kernel(const _Float16* __restrict__ y, const float* __restrict__ scale,
                              const float* __restrict__ shift, const int* __restrict__ seg,
                              float* __restrict__ xout, long rows, int NC) {
  long idx = (long)blockIdx.x * blockDim.x + threadIdx.x;
  if (idx < rows * (long)NC) {
    long r = idx / NC; int c = (int)(idx - r * NC);
    float v = (float)y[idx] * scale[c] + shift[c];
    v = v > 0.f ? v : 0.f;
    atomicMax((unsigned int*)&xout[(long)seg[r] * NC + c], __float_as_uint(v));
  }
}

// ------------------------------------------------- A-fragment run loaders
// Load 8 consecutive K-features starting at k0 for row (run never straddles
// the 64-col x1/x2 boundary: k0 is always a multiple of 8).
template<int MODE, int K>
__device__ __forceinline__ void load_run8(float* __restrict__ f, long row, long s, long d,
                                          int k0,
                                          const float* __restrict__ pos,
                                          const _Float16* __restrict__ yprev,
                                          const float* __restrict__ ssc,
                                          const float* __restrict__ ssh,
                                          const float* __restrict__ x1,
                                          const float* __restrict__ x2) {
  if (MODE == 0) {
    if (k0 == 0) {
      float xi0 = pos[d * 3 + 0], xi1 = pos[d * 3 + 1], xi2 = pos[d * 3 + 2];
      float xj0 = pos[s * 3 + 0], xj1 = pos[s * 3 + 1], xj2 = pos[s * 3 + 2];
      f[0] = xi0; f[1] = xi1; f[2] = xi2;
      f[3] = xj0 - xi0; f[4] = xj1 - xi1; f[5] = xj2 - xi2;
      f[6] = 0.f; f[7] = 0.f;
    } else {
#pragma unroll
      for (int e = 0; e < 8; ++e) f[e] = 0.f;
    }
  } else if (MODE == 1) {
    v8h h = *reinterpret_cast<const v8h*>(yprev + row * K + k0);
#pragma unroll
    for (int e = 0; e < 8; ++e) {
      float t = (float)h[e] * ssc[k0 + e] + ssh[k0 + e];
      f[e] = t > 0.f ? t : 0.f;
    }
  } else if (MODE == 2) {
    if (k0 < 64) {
      v4f a0 = *reinterpret_cast<const v4f*>(x1 + d * 64 + k0);
      v4f a1 = *reinterpret_cast<const v4f*>(x1 + d * 64 + k0 + 4);
#pragma unroll
      for (int e = 0; e < 4; ++e) { f[e] = a0[e]; f[4 + e] = a1[e]; }
    } else {
      int kk = k0 - 64;
      v4f p0 = *reinterpret_cast<const v4f*>(x1 + s * 64 + kk);
      v4f p1 = *reinterpret_cast<const v4f*>(x1 + s * 64 + kk + 4);
      v4f q0 = *reinterpret_cast<const v4f*>(x1 + d * 64 + kk);
      v4f q1 = *reinterpret_cast<const v4f*>(x1 + d * 64 + kk + 4);
#pragma unroll
      for (int e = 0; e < 4; ++e) { f[e] = p0[e] - q0[e]; f[4 + e] = p1[e] - q1[e]; }
    }
  } else {  // MODE 3
    const float* base = (k0 < 64) ? (x1 + row * 64 + k0) : (x2 + row * 64 + (k0 - 64));
    v4f a0 = *reinterpret_cast<const v4f*>(base);
    v4f a1 = *reinterpret_cast<const v4f*>(base + 4);
#pragma unroll
    for (int e = 0; e < 4; ++e) { f[e] = a0[e]; f[4 + e] = a1[e]; }
  }
}

// ----------------------------------------------------- fused GEMM + BN stats
// Y = A @ Wp^T + bias; stores Y f16 via LDS-staged coalesced b128 stores;
// accumulates per-channel sum/sumsq (LDS reduce -> global atomics).
template<int MODE, int K, int NC>
__global__ void __launch_bounds__(256)
gemm_bn_kernel(const float* __restrict__ pos,
               const int* __restrict__ srcI, const int* __restrict__ dstI,
               const _Float16* __restrict__ yprev,
               const float* __restrict__ scale, const float* __restrict__ shift,
               const float* __restrict__ x1, const float* __restrict__ x2,
               const _Float16* __restrict__ Wp,   // [NC][K] f16
               const float* __restrict__ bias,
               _Float16* __restrict__ yout,
               float* __restrict__ gsum, float* __restrict__ gsumsq,
               long nrows) {
  constexpr int NT = NC / 16;
  constexpr int KS = K / 32;

  __shared__ _Float16 tbuf[WPB][16][NC];   // per-wave output tile staging
  __shared__ float lsum[NC];
  __shared__ float lsq[NC];
  __shared__ float ssc[K];                 // BN scale/shift for MODE 1
  __shared__ float ssh[K];

  if (MODE == 1) {
    for (int i = threadIdx.x; i < K; i += blockDim.x) { ssc[i] = scale[i]; ssh[i] = shift[i]; }
  }
  for (int i = threadIdx.x; i < NC; i += blockDim.x) { lsum[i] = 0.f; lsq[i] = 0.f; }
  __syncthreads();

  const int lane = threadIdx.x & 31;
  const int wave = threadIdx.x >> 5;
  const long tile = (long)blockIdx.x * WPB + wave;
  const long row0 = tile * 16;
  const long row  = row0 + (lane & 15);
  const bool rvalid = row < nrows;

  long s = 0, d = 0;
  if ((MODE == 0 || MODE == 2) && rvalid) { s = srcI[row]; d = dstI[row]; }

  const int kbase = (lane >> 4) * 16;  // A-fragment K sub-offset (contiguous order k = 16*(lane/16) + s, same as B)
  v8f acc[NT] = {};

  for (int ks = 0; ks < KS; ++ks) {
    // ---- A fragment: 16x32 f16 = two 8-wide runs per lane
    const int k0 = ks * 32 + kbase;
    float f1[8], f2[8];
    if (rvalid) {
      load_run8<MODE, K>(f1, row, s, d, k0,      pos, yprev, ssc, ssh, x1, x2);
      load_run8<MODE, K>(f2, row, s, d, k0 + 8, pos, yprev, ssc, ssh, x1, x2);
    } else {
#pragma unroll
      for (int e = 0; e < 8; ++e) { f1[e] = 0.f; f2[e] = 0.f; }
    }
    v16h a;
#pragma unroll
    for (int e = 0; e < 8; ++e) { a[e] = (_Float16)f1[e]; a[8 + e] = (_Float16)f2[e]; }

    // ---- B fragments: 32x16 f16; lane's 16 halves contiguous in Wp[n][K]
    const int kb = ks * 32 + ((lane >= 16) ? 16 : 0);
#pragma unroll
    for (int nt = 0; nt < NT; ++nt) {
      const int n = nt * 16 + (lane & 15);
      v16h b = *reinterpret_cast<const v16h*>(Wp + n * K + kb);
      acc[nt] = __builtin_amdgcn_wmma_f32_16x16x32_f16(
          false, a, false, b, (short)0, acc[nt], false, false);
    }
  }

  // ---- epilogue: bias, stats, stage tile into LDS
#pragma unroll
  for (int nt = 0; nt < NT; ++nt) {
    const int col = nt * 16 + (lane & 15);
    const float bs = bias[col];
    float ps = 0.f, pq = 0.f;
#pragma unroll
    for (int r = 0; r < 8; ++r) {
      const int mrow = r + ((lane >= 16) ? 8 : 0);    // C/D layout: V r -> M=r / r+8
      float v = acc[nt][r] + bs;
      tbuf[wave][mrow][col] = (_Float16)v;
      if (row0 + mrow < nrows) { ps += v; pq += v * v; }
    }
    atomicAdd(&lsum[col], ps);
    atomicAdd(&lsq[col], pq);
  }
  __syncthreads();

  // ---- coalesced b128 writeback of the 16 x NC wave tile
  constexpr int CHUNKS = NC / 16;                     // 8-half chunks per lane
#pragma unroll
  for (int c = 0; c < CHUNKS; ++c) {
    const int h   = (c * 32 + lane) * 8;              // half index within tile
    const int mr  = h / NC;
    const int col = h - mr * NC;
    if (row0 + mr < nrows) {
      *reinterpret_cast<v8h*>(yout + (row0 + mr) * NC + col) =
          *reinterpret_cast<const v8h*>(&tbuf[wave][mr][col]);
    }
  }
  for (int i = threadIdx.x; i < NC; i += blockDim.x) {
    atomicAdd(&gsum[i],   lsum[i]);
    atomicAdd(&gsumsq[i], lsq[i]);
  }
}

// --------------------------------------------------- tiny 64-row MLP layers
__global__ void tiny_bn_kernel(const float* __restrict__ in,   // [64,128]
                               const float* __restrict__ W,    // [128,128]
                               const float* __restrict__ b,
                               const float* __restrict__ g, const float* __restrict__ be,
                               float* __restrict__ out) {
  __shared__ float y[64 * 128];
  __shared__ float sc[128], sh[128];
  for (int o = threadIdx.x; o < 64 * 128; o += blockDim.x) {
    int r = o >> 7, c = o & 127;
    float s = b[c];
    for (int k = 0; k < 128; ++k) s += in[r * 128 + k] * W[c * 128 + k];
    y[o] = s;
  }
  __syncthreads();
  if (threadIdx.x < 128) {
    int c = threadIdx.x;
    float s = 0.f, q = 0.f;
    for (int r = 0; r < 64; ++r) { float v = y[r * 128 + c]; s += v; q += v * v; }
    float m = s * (1.f / 64.f);
    float var = q * (1.f / 64.f) - m * m;
    float rs = rsqrtf(var + 1e-5f);
    float a = g[c] * rs;
    sc[c] = a; sh[c] = be[c] - m * a;
  }
  __syncthreads();
  for (int o = threadIdx.x; o < 64 * 128; o += blockDim.x) {
    int c = o & 127;
    float v = y[o] * sc[c] + sh[c];
    out[o] = v > 0.f ? v : 0.f;
  }
}

// logits + log_softmax, one thread per graph
__global__ void head_kernel(const float* __restrict__ in,   // [64,128]
                            const float* __restrict__ W8,   // [40,128]
                            const float* __restrict__ b8,
                            float* __restrict__ out) {      // [64,40]
  int r = threadIdx.x;
  if (r < 64) {
    float logits[40];
    float mx = -3.4e38f;
    for (int c = 0; c < 40; ++c) {
      float s = b8[c];
      for (int k = 0; k < 128; ++k) s += in[r * 128 + k] * W8[c * 128 + k];
      logits[c] = s;
      mx = fmaxf(mx, s);
    }
    float se = 0.f;
    for (int c = 0; c < 40; ++c) se += expf(logits[c] - mx);
    float lse = mx + logf(se);
    for (int c = 0; c < 40; ++c) out[r * 40 + c] = logits[c] - lse;
  }
}

// ---------------------------------------------------------------- launcher
static inline unsigned cdivu(long a, long b) { return (unsigned)((a + b - 1) / b); }

extern "C" void kernel_launch(void* const* d_in, const int* in_sizes, int n_in,
                              void* d_out, int out_size, void* d_ws, size_t ws_size,
                              hipStream_t stream) {
  (void)n_in; (void)out_size; (void)ws_size;
  const long N = in_sizes[0] / 3;
  const long E = in_sizes[1] / 2;
  const int  B = 64;

  const float* pos   = (const float*)d_in[0];
  const int*   ei    = (const int*)d_in[1];
  const int*   srcI  = ei;
  const int*   dstI  = ei + E;
  const int*   batch = (const int*)d_in[2];

  const float *W[9], *bb[9], *gg[9], *bbe[9];
  int idx = 3;
  for (int i = 1; i <= 8; ++i) {
    W[i]  = (const float*)d_in[idx++];
    bb[i] = (const float*)d_in[idx++];
    if (i < 8) { gg[i] = (const float*)d_in[idx++]; bbe[i] = (const float*)d_in[idx++]; }
    else       { gg[i] = nullptr; bbe[i] = nullptr; }
  }

  // workspace layout (256B aligned)
  char* ws = (char*)d_ws;
  size_t off = 0;
  auto take = [&](size_t bytes) -> void* {
    void* p = ws + off; off = (off + bytes + 255) & ~(size_t)255; return p;
  };
  _Float16* yA     = (_Float16*)take((size_t)E * 64 * 2);
  _Float16* yB     = (_Float16*)take((size_t)E * 64 * 2);
  float*    x1     = (float*)   take((size_t)N * 64 * 4);
  float*    x2     = (float*)   take((size_t)N * 64 * 4);
  _Float16* y5     = (_Float16*)take((size_t)N * 128 * 2);
  float*    pooled = (float*)   take((size_t)B * 128 * 4);
  float*    t6     = (float*)   take((size_t)B * 128 * 4);
  float*    t7     = (float*)   take((size_t)B * 128 * 4);
  _Float16* W1p    = (_Float16*)take(64 * 32 * 2);
  _Float16* W2p    = (_Float16*)take(64 * 64 * 2);
  _Float16* W3p    = (_Float16*)take(64 * 64 * 2);
  _Float16* W4p    = (_Float16*)take(64 * 128 * 2);
  _Float16* W5p    = (_Float16*)take(128 * 128 * 2);
  float*    stats  = (float*)   take(5 * 4 * 128 * 4);
  auto S  = [&](int s) { return stats + s * 512; };        // sum
  auto Q  = [&](int s) { return stats + s * 512 + 128; };  // sumsq
  auto SC = [&](int s) { return stats + s * 512 + 256; };  // scale
  auto SH = [&](int s) { return stats + s * 512 + 384; };  // shift

  dim3 blk(256);

  // weight prep (f32 [out,in] -> f16 [out,Kpad])
  wprep_kernel<<<cdivu(64 * 32, 256), blk, 0, stream>>>(W[1], W1p, 6, 32, 64);
  wprep_kernel<<<cdivu(64 * 64, 256), blk, 0, stream>>>(W[2], W2p, 64, 64, 64);
  wprep_kernel<<<cdivu(64 * 64, 256), blk, 0, stream>>>(W[3], W3p, 64, 64, 64);
  wprep_kernel<<<cdivu(64 * 128, 256), blk, 0, stream>>>(W[4], W4p, 128, 128, 64);
  wprep_kernel<<<cdivu(128 * 128, 256), blk, 0, stream>>>(W[5], W5p, 128, 128, 128);

  // zero aggregation targets + stats
  zero_f32_kernel<<<cdivu(N * 64, 256), blk, 0, stream>>>(x1, N * 64);
  zero_f32_kernel<<<cdivu(N * 64, 256), blk, 0, stream>>>(x2, N * 64);
  zero_f32_kernel<<<cdivu(B * 128, 256), blk, 0, stream>>>(pooled, (long)B * 128);
  zero_f32_kernel<<<cdivu(5 * 4 * 128, 256), blk, 0, stream>>>(stats, 5 * 4 * 128);

  const unsigned eblocks = cdivu((E + 15) / 16, WPB);
  const unsigned nblocks = cdivu((N + 15) / 16, WPB);
  const float invE = 1.0f / (float)E;
  const float invN = 1.0f / (float)N;

  // conv1 layer1: edge features (6) -> 64
  gemm_bn_kernel<0, 32, 64><<<eblocks, blk, 0, stream>>>(
      pos, srcI, dstI, nullptr, nullptr, nullptr, nullptr, nullptr,
      W1p, bb[1], yA, S(0), Q(0), E);
  finalize_kernel<<<1, 128, 0, stream>>>(S(0), Q(0), gg[1], bbe[1], SC(0), SH(0), invE, 64);

  // conv1 layer2: 64 -> 64 (bn+relu of yA fused into A-load)
  gemm_bn_kernel<1, 64, 64><<<eblocks, blk, 0, stream>>>(
      nullptr, nullptr, nullptr, yA, SC(0), SH(0), nullptr, nullptr,
      W2p, bb[2], yB, S(1), Q(1), E);
  finalize_kernel<<<1, 128, 0, stream>>>(S(1), Q(1), gg[2], bbe[2], SC(1), SH(1), invE, 64);

  // conv1 layer3: 64 -> 64
  gemm_bn_kernel<1, 64, 64><<<eblocks, blk, 0, stream>>>(
      nullptr, nullptr, nullptr, yB, SC(1), SH(1), nullptr, nullptr,
      W3p, bb[3], yA, S(2), Q(2), E);
  finalize_kernel<<<1, 128, 0, stream>>>(S(2), Q(2), gg[3], bbe[3], SC(2), SH(2), invE, 64);

  // segment-max over dst -> x1 [N,64]
  maxagg_kernel<<<cdivu(E * 64, 256), blk, 0, stream>>>(yA, SC(2), SH(2), dstI, x1, E, 64);

  // conv2: edge features from x1 (128) -> 64
  gemm_bn_kernel<2, 128, 64><<<eblocks, blk, 0, stream>>>(
      nullptr, srcI, dstI, nullptr, nullptr, nullptr, x1, nullptr,
      W4p, bb[4], yB, S(3), Q(3), E);
  finalize_kernel<<<1, 128, 0, stream>>>(S(3), Q(3), gg[4], bbe[4], SC(3), SH(3), invE, 64);

  // segment-max -> x2 [N,64]
  maxagg_kernel<<<cdivu(E * 64, 256), blk, 0, stream>>>(yB, SC(3), SH(3), dstI, x2, E, 64);

  // node GEMM: cat[x1,x2] (128) -> 128
  gemm_bn_kernel<3, 128, 128><<<nblocks, blk, 0, stream>>>(
      nullptr, nullptr, nullptr, nullptr, nullptr, nullptr, x1, x2,
      W5p, bb[5], y5, S(4), Q(4), N);
  finalize_kernel<<<1, 128, 0, stream>>>(S(4), Q(4), gg[5], bbe[5], SC(4), SH(4), invN, 128);

  // graph max-pool -> pooled [64,128]
  maxagg_kernel<<<cdivu(N * 128, 256), blk, 0, stream>>>(y5, SC(4), SH(4), batch, pooled, N, 128);

  // tiny head
  tiny_bn_kernel<<<1, 256, 0, stream>>>(pooled, W[6], bb[6], gg[6], bbe[6], t6);
  tiny_bn_kernel<<<1, 256, 0, stream>>>(t6, W[7], bb[7], gg[7], bbe[7], t7);
  head_kernel<<<1, 64, 0, stream>>>(t7, W[8], bb[8], (float*)d_out);
}
